// GMNLayerX_Pooling2_28432683499989
// MI455X (gfx1250) — matrix operand order repaired and checked by reading, not compiled, not run on hardware
//
#include <hip/hip_runtime.h>
#include <hip/hip_bf16.h>

// ---------------------------------------------------------------------------
// Problem constants (from reference)
// ---------------------------------------------------------------------------
#define TT 10
#define NN 2048
#define EE 32768
#define HH 256

typedef __attribute__((ext_vector_type(16))) __bf16 v16bf;
typedef __attribute__((ext_vector_type(8)))  float  v8f;
typedef __attribute__((ext_vector_type(4)))  unsigned int u32x4;

union Frag16 { u32x4 u[2]; v16bf v; };

// ---------------------------------------------------------------------------
// A-fragment load (16x32 bf16, M-rows striped per ISA layout) from LDS tile.
// lane<16 : row=mbase+lane,    K = kt*32 + {0..7, 16..23}
// lane>=16: row=mbase+lane-16, K = kt*32 + {8..15, 24..31}
// ---------------------------------------------------------------------------
__device__ __forceinline__ v16bf load_a_frag(const __bf16* base, int strideElems,
                                             int mbase, int kt, int lane) {
    int row  = mbase + (lane & 15);
    int koff = kt * 32 + ((lane >> 4) << 3);
    const __bf16* p = base + row * strideElems + koff;
    Frag16 f;
    f.u[0] = *(const u32x4*)(p);        // K +0..7
    f.u[1] = *(const u32x4*)(p + 16);   // K +16..23
    return f.v;
}

// ---------------------------------------------------------------------------
// B-fragment load from fragment-packed weights in global memory (L2 resident).
// Packed layout: [nt][kt][lane][16 bf16 contiguous] -> one 32B load per lane.
// ---------------------------------------------------------------------------
__device__ __forceinline__ v16bf load_b_frag(const __bf16* pack, int nt, int kt,
                                             int Kt, int lane) {
    const __bf16* p = pack + ((((size_t)nt * Kt + kt) * 32 + lane) << 4);
    Frag16 f;
    f.u[0] = *(const u32x4*)(p);
    f.u[1] = *(const u32x4*)(p + 8);
    return f.v;
}

__device__ __forceinline__ void atomicAddF32(float* p, float v) {
    unsafeAtomicAdd(p, v);   // lowers to global_atomic_add_f32 on gfx1250
}

// ---------------------------------------------------------------------------
// Pack fp32 weight W[Kreal x 256] (row-major fan_in x fan_out) into bf16
// B-fragment order, zero-padding K to Kpad (multiple of 32).
// ---------------------------------------------------------------------------
__global__ void pack_weight(const float* __restrict__ W, __bf16* __restrict__ out,
                            int Kreal, int Kpad) {
    const int Kt = Kpad / 32;
    const int total = 16 * Kt * 512;
    for (int i = blockIdx.x * blockDim.x + threadIdx.x; i < total;
         i += gridDim.x * blockDim.x) {
        int q    = i & 15;
        int lane = (i >> 4) & 31;
        int kt   = (i >> 9) % Kt;
        int nt   = (i >> 9) / Kt;
        int col  = nt * 16 + (lane & 15);
        int k    = kt * 32 + ((lane >> 4) << 3) + (q & 7) + ((q >> 3) << 4);
        float v  = (k < Kreal) ? W[(size_t)k * 256 + col] : 0.0f;
        out[i]   = (__bf16)v;
    }
}

__global__ void zero_f32(float* __restrict__ p, long n) {
    for (long i = blockIdx.x * (long)blockDim.x + threadIdx.x; i < n;
         i += (long)gridDim.x * blockDim.x) p[i] = 0.0f;
}

__global__ void count_edges(const int* __restrict__ ei, float* __restrict__ counts) {
    int i = blockIdx.x * blockDim.x + threadIdx.x;
    if (i < EE) atomicAddF32(&counts[ei[i]], 1.0f);
}

// ---------------------------------------------------------------------------
// Shared GEMM stage: [64 x K] (LDS, bf16) @ packed W [K x 256] -> LDS bf16.
// Each wave owns 2 ntiles and 4 mtiles with ALL accumulators held live
// (acc0[4] + acc1[4] = 64 VGPRs) and a single ROLLED kt loop:
//   per kt: 2 B loads (each feeds 4 WMMAs) + 4 A loads (each feeds 2 WMMAs)
// Rolled loop keeps live set ~90 VGPRs -> no spills, nothing to hoist.
// ---------------------------------------------------------------------------
__device__ __forceinline__ void gemm_stage64(
    const __bf16* __restrict__ Alds, int strideA, int Kt,
    const __bf16* __restrict__ Bpack, const float* __restrict__ bias,
    __bf16* __restrict__ Out, int wave, int lane)
{
    const int nt0 = wave * 2;
    const int nt1 = nt0 + 1;
    v8f acc0[4], acc1[4];
    #pragma unroll
    for (int mt = 0; mt < 4; ++mt) {
        acc0[mt] = (v8f){0.f,0.f,0.f,0.f,0.f,0.f,0.f,0.f};
        acc1[mt] = (v8f){0.f,0.f,0.f,0.f,0.f,0.f,0.f,0.f};
    }
    #pragma unroll 1
    for (int kt = 0; kt < Kt; ++kt) {
        v16bf b0 = load_b_frag(Bpack, nt0, kt, Kt, lane);
        v16bf b1 = load_b_frag(Bpack, nt1, kt, Kt, lane);
        #pragma unroll
        for (int mt = 0; mt < 4; ++mt) {
            v16bf a = load_a_frag(Alds, strideA, mt * 16, kt, lane);
            acc0[mt] = __builtin_amdgcn_wmma_f32_16x16x32_bf16(
                false, a, false, b0, (short)0, acc0[mt], false, false);
            acc1[mt] = __builtin_amdgcn_wmma_f32_16x16x32_bf16(
                false, a, false, b1, (short)0, acc1[mt], false, false);
        }
    }
    const int colG0 = lane & 15;
    const int rsub  = (lane >> 4) << 3;
    const int col0  = nt0 * 16 + colG0;
    const int col1  = nt1 * 16 + colG0;
    const float bs0 = bias[col0];
    const float bs1 = bias[col1];
    #pragma unroll
    for (int mt = 0; mt < 4; ++mt) {
        int rb = mt * 16 + rsub;
        #pragma unroll
        for (int r = 0; r < 8; ++r) {
            Out[(rb + r) * HH + col0] = (__bf16)fmaxf(acc0[mt][r] + bs0, 0.0f);
            Out[(rb + r) * HH + col1] = (__bf16)fmaxf(acc1[mt][r] + bs1, 0.0f);
        }
    }
}

// ---------------------------------------------------------------------------
// Fused edge kernel: 64 edges per block, one timestep per block.
//   radial -> e1=relu(edge_in@We1+be1) -> e=relu(e1@We2+be2)
//   c1=relu(e@Wc1+bc1) -> m=c1@Wc2 ; scatter: agg+=e ; f+=coord_diff*m
// ---------------------------------------------------------------------------
#define ETILE 64
#define KPAD1 544   // 2H+16=528 padded to multiple of 32

__global__ __launch_bounds__(256) void edge_kernel(
    const float* __restrict__ x,  const float* __restrict__ h,
    const int*   __restrict__ ei,
    const __bf16* __restrict__ We1p, const float* __restrict__ be1,
    const __bf16* __restrict__ We2p, const float* __restrict__ be2,
    const __bf16* __restrict__ Wc1p, const float* __restrict__ bc1,
    const float* __restrict__ Wc2,
    float* __restrict__ agg, float* __restrict__ fbuf)
{
    __shared__ __align__(16) char smem[105984];
    __bf16* Ain  = (__bf16*)smem;                       // 64 x 544 edge_in (bf16)
    __bf16* bufB = (__bf16*)smem;                       // alias: 64 x 256 (e)
    __bf16* bufA = (__bf16*)(smem + 69632);             // 64 x 256 (e1 / c1)
    float*  sCd  = (float*)(smem + 102400);             // 64 x 12 coord_diff
    int*    sRow = (int*)(smem + 105472);
    int*    sCol = (int*)(smem + 105728);

    const int tilesPerT = EE / ETILE;                   // 512
    const int t     = blockIdx.x / tilesPerT;
    const int ebase = (blockIdx.x % tilesPerT) * ETILE;
    const int tid   = threadIdx.x;
    const int lane  = tid & 31;
    const int wave  = tid >> 5;

    if (tid < ETILE) {
        sRow[tid] = ei[ebase + tid];
        sCol[tid] = ei[EE + ebase + tid];
    }
    __syncthreads();

    // ---- gather h[row], h[col] into Ain (bf16) ----
    for (int idx = tid; idx < ETILE * HH; idx += 256) {
        int e = idx >> 8, c = idx & 255;
        float hr = h[((size_t)t * NN + sRow[e]) * HH + c];
        float hc = h[((size_t)t * NN + sCol[e]) * HH + c];
        Ain[e * KPAD1 + c]       = (__bf16)hr;
        Ain[e * KPAD1 + 256 + c] = (__bf16)hc;
    }
    // ---- coord_diff + normalized gram (radial) + K padding ----
    if (tid < ETILE) {
        int e = tid;
        const float* xr = x + ((size_t)t * NN + sRow[e]) * 12;
        const float* xc = x + ((size_t)t * NN + sCol[e]) * 12;
        float cd[12];
        #pragma unroll
        for (int j = 0; j < 12; ++j) { cd[j] = xr[j] - xc[j]; sCd[e * 12 + j] = cd[j]; }
        float g[16]; float ss = 0.0f;
        #pragma unroll
        for (int j = 0; j < 4; ++j)
            #pragma unroll
            for (int k = 0; k < 4; ++k) {
                float v = cd[j*3]*cd[k*3] + cd[j*3+1]*cd[k*3+1] + cd[j*3+2]*cd[k*3+2];
                g[j*4+k] = v; ss += v * v;
            }
        float inv = 1.0f / fmaxf(sqrtf(ss), 1e-12f);
        #pragma unroll
        for (int j = 0; j < 16; ++j) Ain[e * KPAD1 + 512 + j] = (__bf16)(g[j] * inv);
        #pragma unroll
        for (int j = 528; j < KPAD1; ++j) Ain[e * KPAD1 + j] = (__bf16)0.0f;
    }
    __syncthreads();

    // ---- Stage 1: e1 = relu(edge_in @ We1 + be1) ----
    gemm_stage64(Ain, KPAD1, KPAD1 / 32, We1p, be1, bufA, wave, lane);
    __syncthreads();
    // ---- Stage 2: e = relu(e1 @ We2 + be2) ----
    gemm_stage64(bufA, HH, 8, We2p, be2, bufB, wave, lane);
    __syncthreads();
    // ---- Stage 3: c1 = relu(e @ Wc1 + bc1) ----
    gemm_stage64(bufB, HH, 8, Wc1p, bc1, bufA, wave, lane);
    __syncthreads();

    // ---- m = c1 @ Wc2 ; trans scatter (12 atomics/edge) ----
    if (tid < ETILE) {
        float mm = 0.0f;
        for (int c = 0; c < HH; ++c) mm += (float)bufA[tid * HH + c] * Wc2[c];
        float* fdst = fbuf + ((size_t)t * NN + sRow[tid]) * 12;
        #pragma unroll
        for (int j = 0; j < 12; ++j) atomicAddF32(&fdst[j], sCd[tid * 12 + j] * mm);
    }
    // ---- agg scatter: agg[t, row, :] += e ----
    for (int idx = tid; idx < ETILE * HH; idx += 256) {
        int e = idx >> 8, c = idx & 255;
        atomicAddF32(&agg[((size_t)t * NN + sRow[e]) * HH + c], (float)bufB[e * HH + c]);
    }
}

// ---------------------------------------------------------------------------
// Fused node kernel: 32 nodes per block, one timestep per block.
//   node_in=[others,h,agg] -> n1=relu(@Wn1+bn1) -> h_out = h + n1@Wn2 + bn2
//   coord_out = x + f/max(counts,1)
// Each wave: 2 ntiles x 2 mtiles, all accumulators live, rolled kt loop.
// ---------------------------------------------------------------------------
#define KPADN 768

__global__ __launch_bounds__(256) void node_kernel(
    const float* __restrict__ x, const float* __restrict__ h,
    const float* __restrict__ others,
    const __bf16* __restrict__ Wn1p, const float* __restrict__ bn1,
    const __bf16* __restrict__ Wn2p, const float* __restrict__ bn2,
    const float* __restrict__ agg, const float* __restrict__ fbuf,
    const float* __restrict__ counts,
    float* __restrict__ hout, float* __restrict__ coordout)
{
    __shared__ __align__(16) char smem[65536];
    __bf16* nin  = (__bf16*)smem;               // 32 x 768
    __bf16* nbuf = (__bf16*)(smem + 49152);     // 32 x 256

    const int tilesPerT = NN / 32;              // 64
    const int t     = blockIdx.x / tilesPerT;
    const int nbase = (blockIdx.x % tilesPerT) * 32;
    const int tid   = threadIdx.x;
    const int lane  = tid & 31;
    const int wave  = tid >> 5;

    // ---- build node_in = [others | h | agg] as bf16 ----
    for (int idx = tid; idx < 32 * HH; idx += 256) {
        int n = idx >> 8, c = idx & 255;
        size_t row = (size_t)t * NN + nbase + n;
        nin[n * KPADN + c]       = (__bf16)others[row * HH + c];
        nin[n * KPADN + 256 + c] = (__bf16)h[row * HH + c];
        nin[n * KPADN + 512 + c] = (__bf16)agg[row * HH + c];
    }
    __syncthreads();

    const int colG0 = lane & 15;
    const int rsub  = (lane >> 4) << 3;
    const int nt0   = wave * 2;
    const int nt1   = nt0 + 1;

    // ---- Stage 1: n1 = relu(node_in @ Wn1 + bn1) ----
    {
        v8f acc0[2], acc1[2];
        #pragma unroll
        for (int mt = 0; mt < 2; ++mt) {
            acc0[mt] = (v8f){0.f,0.f,0.f,0.f,0.f,0.f,0.f,0.f};
            acc1[mt] = (v8f){0.f,0.f,0.f,0.f,0.f,0.f,0.f,0.f};
        }
        #pragma unroll 1
        for (int kt = 0; kt < KPADN / 32; ++kt) {
            v16bf b0 = load_b_frag(Wn1p, nt0, kt, KPADN / 32, lane);
            v16bf b1 = load_b_frag(Wn1p, nt1, kt, KPADN / 32, lane);
            #pragma unroll
            for (int mt = 0; mt < 2; ++mt) {
                v16bf a = load_a_frag(nin, KPADN, mt * 16, kt, lane);
                acc0[mt] = __builtin_amdgcn_wmma_f32_16x16x32_bf16(
                    false, a, false, b0, (short)0, acc0[mt], false, false);
                acc1[mt] = __builtin_amdgcn_wmma_f32_16x16x32_bf16(
                    false, a, false, b1, (short)0, acc1[mt], false, false);
            }
        }
        int col0 = nt0 * 16 + colG0, col1 = nt1 * 16 + colG0;
        float bs0 = bn1[col0], bs1 = bn1[col1];
        #pragma unroll
        for (int mt = 0; mt < 2; ++mt) {
            int rb = mt * 16 + rsub;
            #pragma unroll
            for (int r = 0; r < 8; ++r) {
                nbuf[(rb + r) * HH + col0] = (__bf16)fmaxf(acc0[mt][r] + bs0, 0.0f);
                nbuf[(rb + r) * HH + col1] = (__bf16)fmaxf(acc1[mt][r] + bs1, 0.0f);
            }
        }
    }
    __syncthreads();

    // ---- Stage 2: h_out = h + n1 @ Wn2 + bn2 ----
    {
        v8f acc0[2], acc1[2];
        #pragma unroll
        for (int mt = 0; mt < 2; ++mt) {
            acc0[mt] = (v8f){0.f,0.f,0.f,0.f,0.f,0.f,0.f,0.f};
            acc1[mt] = (v8f){0.f,0.f,0.f,0.f,0.f,0.f,0.f,0.f};
        }
        #pragma unroll 1
        for (int kt = 0; kt < 8; ++kt) {
            v16bf b0 = load_b_frag(Wn2p, nt0, kt, 8, lane);
            v16bf b1 = load_b_frag(Wn2p, nt1, kt, 8, lane);
            #pragma unroll
            for (int mt = 0; mt < 2; ++mt) {
                v16bf a = load_a_frag(nbuf, HH, mt * 16, kt, lane);
                acc0[mt] = __builtin_amdgcn_wmma_f32_16x16x32_bf16(
                    false, a, false, b0, (short)0, acc0[mt], false, false);
                acc1[mt] = __builtin_amdgcn_wmma_f32_16x16x32_bf16(
                    false, a, false, b1, (short)0, acc1[mt], false, false);
            }
        }
        int col0 = nt0 * 16 + colG0, col1 = nt1 * 16 + colG0;
        float bs0 = bn2[col0], bs1 = bn2[col1];
        #pragma unroll
        for (int mt = 0; mt < 2; ++mt) {
            int rb = mt * 16 + rsub;
            #pragma unroll
            for (int r = 0; r < 8; ++r) {
                size_t ix0 = ((size_t)t * NN + nbase + rb + r) * HH + col0;
                size_t ix1 = ((size_t)t * NN + nbase + rb + r) * HH + col1;
                hout[ix0] = acc0[mt][r] + bs0 + h[ix0];
                hout[ix1] = acc1[mt][r] + bs1 + h[ix1];
            }
        }
    }

    // ---- coord epilogue: coord_out = x + f / max(counts,1) ----
    for (int idx = tid; idx < 32 * 12; idx += 256) {
        int n = idx / 12, j = idx % 12;
        size_t ix = ((size_t)t * NN + nbase + n) * 12 + j;
        float cnt = fmaxf(counts[nbase + n], 1.0f);
        coordout[ix] = x[ix] + fbuf[ix] / cnt;
    }
}

// ---------------------------------------------------------------------------
// Launch
// ---------------------------------------------------------------------------
extern "C" void kernel_launch(void* const* d_in, const int* in_sizes, int n_in,
                              void* d_out, int out_size, void* d_ws, size_t ws_size,
                              hipStream_t stream) {
    const float* x      = (const float*)d_in[0];
    const float* h      = (const float*)d_in[1];
    const float* others = (const float*)d_in[2];
    const float* We1    = (const float*)d_in[3];
    const float* be1    = (const float*)d_in[4];
    const float* We2    = (const float*)d_in[5];
    const float* be2    = (const float*)d_in[6];
    const float* Wn1    = (const float*)d_in[7];
    const float* bn1    = (const float*)d_in[8];
    const float* Wn2    = (const float*)d_in[9];
    const float* bn2    = (const float*)d_in[10];
    const float* Wc1    = (const float*)d_in[11];
    const float* bc1    = (const float*)d_in[12];
    const float* Wc2    = (const float*)d_in[13];
    const int*   ei     = (const int*)d_in[14];

    char* ws = (char*)d_ws;
    __bf16* We1p  = (__bf16*)(ws);                       // 544*256*2 = 278528
    __bf16* We2p  = (__bf16*)(ws + 278528);              // 131072
    __bf16* Wc1p  = (__bf16*)(ws + 409600);              // 131072
    __bf16* Wn1p  = (__bf16*)(ws + 540672);              // 768*256*2 = 393216
    __bf16* Wn2p  = (__bf16*)(ws + 933888);              // 131072
    float* agg    = (float*)(ws + 1048576);              // T*N*H f32 = 20971520
    float* fbuf   = (float*)(ws + 1048576 + 20971520);   // T*N*12 f32 = 983040
    float* counts = (float*)(ws + 1048576 + 20971520 + 983040); // N f32

    float* hout     = (float*)d_out;
    float* coordout = hout + (size_t)TT * NN * HH;

    // per-call weight packing (deterministic, idempotent)
    pack_weight<<<256, 256, 0, stream>>>(We1, We1p, 528, KPAD1);
    pack_weight<<<256, 256, 0, stream>>>(We2, We2p, 256, 256);
    pack_weight<<<256, 256, 0, stream>>>(Wc1, Wc1p, 256, 256);
    pack_weight<<<256, 256, 0, stream>>>(Wn1, Wn1p, 768, KPADN);
    pack_weight<<<256, 256, 0, stream>>>(Wn2, Wn2p, 256, 256);

    // zero accumulators (agg + f + counts are contiguous)
    long zcount = (20971520L + 983040L + 8192L) / 4;
    zero_f32<<<1024, 256, 0, stream>>>(agg, zcount);
    count_edges<<<EE / 256, 256, 0, stream>>>(ei, counts);

    edge_kernel<<<TT * (EE / ETILE), 256, 0, stream>>>(
        x, h, ei, We1p, be1, We2p, be2, Wc1p, bc1, Wc2, agg, fbuf);

    node_kernel<<<TT * (NN / 32), 256, 0, stream>>>(
        x, h, others, Wn1p, bn1, Wn2p, bn2, agg, fbuf, counts, hout, coordout);
}
